// OrthoLinear_35811437314863
// MI455X (gfx1250) — compile-verified, matrix operand order checked by reading, not run on hardware
//
#include <hip/hip_runtime.h>
#include <hip/hip_fp16.h>
#include <stdint.h>

// Problem constants (from reference)
#define IN_F   4096
#define OUT_F  4096
#define NROWS  8192           // B*S = 4*2048

typedef __attribute__((ext_vector_type(16))) _Float16 v16h;
typedef __attribute__((ext_vector_type(8)))  _Float16 v8h;
typedef __attribute__((ext_vector_type(8)))  float    v8f;
typedef __attribute__((ext_vector_type(4)))  int      v4i;

// Address-space qualified pointer types for the CDNA5 async-to-LDS builtin,
// whose params are (int4 addrspace(1)*, int4 addrspace(3)*, imm, imm).
typedef __attribute__((address_space(1))) v4i* gv4i_p;
typedef __attribute__((address_space(3))) v4i* lv4i_p;

// ---- CDNA5 async global->LDS copy (ASYNCcnt path), with sync fallback ----
__device__ __forceinline__ void async_copy16(const _Float16* g, _Float16* l) {
#if __has_builtin(__builtin_amdgcn_global_load_async_to_lds_b128)
  __builtin_amdgcn_global_load_async_to_lds_b128((gv4i_p)(void*)g, (lv4i_p)(void*)l, 0, 0);
#else
  *(uint4*)l = *(const uint4*)g;   // fallback: sync copy through VGPRs
#endif
}

__device__ __forceinline__ void wait_async_all() {
#if __has_builtin(__builtin_amdgcn_s_wait_asynccnt)
  __builtin_amdgcn_s_wait_asynccnt(0);
#elif __has_builtin(__builtin_amdgcn_global_load_async_to_lds_b128)
  asm volatile("s_wait_asynccnt 0" ::: "memory");
#endif
}

// ---- Kernel 1: copy base weights (fp16) into workspace so we can fold sparse in ----
__global__ void copy_weights_kernel(const __half* __restrict__ src,
                                    _Float16* __restrict__ dst, int n16) {
  int i = blockIdx.x * blockDim.x + threadIdx.x;   // each thread moves 16 bytes
  if (i < n16) ((uint4*)dst)[i] = ((const uint4*)src)[i];
}

// ---- Kernel 2: scatter-add sparse fp16 outliers into the combined weights ----
// CAS on the containing 32-bit word so duplicate (row,col) pairs accumulate
// (matches .at[rows, cols].add semantics).
__global__ void scatter_nnz_kernel(const int* __restrict__ rows,
                                   const int* __restrict__ cols,
                                   const __half* __restrict__ vals,
                                   _Float16* __restrict__ Wc, int nnz) {
  int i = blockIdx.x * blockDim.x + threadIdx.x;
  if (i >= nnz) return;
  size_t idx = (size_t)rows[i] * IN_F + (size_t)cols[i];
  unsigned int* word = (unsigned int*)((uintptr_t)(Wc + idx) & ~(uintptr_t)3);
  const int hi = (int)(idx & 1);
  const float add = __half2float(vals[i]);
  unsigned int old = *word, assumed;
  do {
    assumed = old;
    unsigned short h = hi ? (unsigned short)(assumed >> 16)
                          : (unsigned short)(assumed & 0xFFFFu);
    float cur = __half2float(__ushort_as_half(h));
    unsigned short nh = __half_as_ushort(__float2half(cur + add));
    unsigned int repl = hi ? ((assumed & 0x0000FFFFu) | ((unsigned int)nh << 16))
                           : ((assumed & 0xFFFF0000u) | (unsigned int)nh);
    old = atomicCAS(word, assumed, repl);
  } while (old != assumed);
}

// ---- Kernel 3: x fp32 -> fp16 once (halves A-side traffic; GEMM is then pure fp16) ----
__global__ void convert_x_kernel(const float* __restrict__ x,
                                 _Float16* __restrict__ x16, int n8) {
  int i = blockIdx.x * blockDim.x + threadIdx.x;   // 8 floats -> 8 halves per thread
  if (i >= n8) return;
  float4 a = ((const float4*)x)[2 * i];
  float4 b = ((const float4*)x)[2 * i + 1];
  v8h o;
  o[0] = (_Float16)a.x; o[1] = (_Float16)a.y; o[2] = (_Float16)a.z; o[3] = (_Float16)a.w;
  o[4] = (_Float16)b.x; o[5] = (_Float16)b.y; o[6] = (_Float16)b.z; o[7] = (_Float16)b.w;
  *(v8h*)(x16 + (size_t)i * 8) = o;
}

// ---- Kernel 4: main WMMA GEMM: out[M,N] = A16[M,K] * W[N,K]^T, fp32 accum ----
// Workgroup tile 128x128, BK=32, double-buffered LDS fed by async-to-LDS.
// 8 waves, each computes 32(M) x 64(N) = 2x4 tiles of v_wmma_f32_16x16x32_f16.
#define BM 128
#define BN 128
#define BK 32
#define LDSB (BK + 8)   // padded row stride in halves: 80B rows, conflict-free frag loads

__global__ __launch_bounds__(256) void gemm_wmma_kernel(
    const _Float16* __restrict__ A16,   // [NROWS][IN_F]
    const _Float16* __restrict__ W,     // [OUT_F][IN_F] combined weights
    float* __restrict__ out)            // [NROWS][OUT_F]
{
  constexpr int K = IN_F;
  __shared__ _Float16 As[2][BM * LDSB];
  __shared__ _Float16 Bs[2][BN * LDSB];

  const int tid = threadIdx.x;
  const int bm0 = blockIdx.y * BM;
  const int bn0 = blockIdx.x * BN;

  const int wave = tid >> 5;
  const int lane = tid & 31;
  const int lr   = lane & 15;   // row/col within 16
  const int lh   = lane >> 4;   // half-wave selector (K sub-chunk)
  const int wm   = (wave & 3) * 32;  // wave M offset inside block tile
  const int wn   = (wave >> 2) * 64; // wave N offset inside block tile

  v8f acc[2][4] = {};

  auto issue_stage = [&](int buf, int k0) {
    const _Float16* ga = A16 + (size_t)bm0 * K + k0;
    const _Float16* gb = W   + (size_t)bn0 * K + k0;
    // 128 rows * 32 halves = 512 x 16B chunks per operand; 2 chunks/thread each
#pragma unroll
    for (int i = 0; i < 2; ++i) {
      int chunk = tid + i * 256;
      int row = chunk >> 2;           // 4 chunks per row
      int cc  = (chunk & 3) * 8;      // half offset within row
      async_copy16(ga + (size_t)row * K + cc, &As[buf][row * LDSB + cc]);
    }
#pragma unroll
    for (int i = 0; i < 2; ++i) {
      int chunk = tid + i * 256;
      int row = chunk >> 2;
      int cc  = (chunk & 3) * 8;
      async_copy16(gb + (size_t)row * K + cc, &Bs[buf][row * LDSB + cc]);
    }
  };

  issue_stage(0, 0);
  constexpr int nstages = K / BK;   // 128

  for (int s = 0; s < nstages; ++s) {
    const int buf = s & 1;
    wait_async_all();     // my stage-s chunks have landed in LDS
    __syncthreads();      // everyone's chunks visible
    if (s + 1 < nstages) issue_stage(buf ^ 1, (s + 1) * BK);  // overlap with compute

    const int c0 = lh * 8;  // A/B fragment K layout: v0..3 = K[c0..c0+7], v4..7 = K[c0+16..c0+23]
    v16h afrag[2];
#pragma unroll
    for (int tm = 0; tm < 2; ++tm) {
      const _Float16* base = &As[buf][(wm + tm * 16 + lr) * LDSB];
      union { v16h v; v8h h[2]; } u;
      u.h[0] = *(const v8h*)(base + c0);
      u.h[1] = *(const v8h*)(base + c0 + 16);
      afrag[tm] = u.v;
    }
    v16h bfrag[4];
#pragma unroll
    for (int tn = 0; tn < 4; ++tn) {
      const _Float16* base = &Bs[buf][(wn + tn * 16 + lr) * LDSB];
      union { v16h v; v8h h[2]; } u;
      u.h[0] = *(const v8h*)(base + c0);
      u.h[1] = *(const v8h*)(base + c0 + 16);
      bfrag[tn] = u.v;
    }
#pragma unroll
    for (int tm = 0; tm < 2; ++tm)
#pragma unroll
      for (int tn = 0; tn < 4; ++tn)
        acc[tm][tn] = __builtin_amdgcn_wmma_f32_16x16x32_f16(
            false, afrag[tm], false, bfrag[tn],
            (short)0, acc[tm][tn], false, false);

    __syncthreads();   // all reads of buf done before it is refilled next iteration
  }

  // Epilogue: C layout — VGPR i: M = i + lh*8 (lanes 0-15 / 16-31), N = lr
#pragma unroll
  for (int tm = 0; tm < 2; ++tm) {
#pragma unroll
    for (int tn = 0; tn < 4; ++tn) {
      const int col = bn0 + wn + tn * 16 + lr;
#pragma unroll
      for (int i = 0; i < 8; ++i) {
        const int row = bm0 + wm + tm * 16 + lh * 8 + i;
        out[(size_t)row * OUT_F + col] = acc[tm][tn][i];
      }
    }
  }
}

extern "C" void kernel_launch(void* const* d_in, const int* in_sizes, int n_in,
                              void* d_out, int out_size, void* d_ws, size_t ws_size,
                              hipStream_t stream) {
  const float*  x    = (const float*)d_in[0];
  const __half* bw   = (const __half*)d_in[1];
  const __half* vals = (const __half*)d_in[2];
  const int*    rows = (const int*)d_in[3];
  const int*    cols = (const int*)d_in[4];
  float* out = (float*)d_out;

  // Workspace layout: [x16 fp16: 64MB][Wc fp16: 32MB]
  _Float16* x16 = (_Float16*)d_ws;
  _Float16* Wc  = (_Float16*)((char*)d_ws + (size_t)NROWS * IN_F * sizeof(_Float16));

  const int nnz = in_sizes[2];

  // 1) Wc = base_weights
  const int n16 = (OUT_F * IN_F) / 8;                    // 16B chunks
  copy_weights_kernel<<<(n16 + 255) / 256, 256, 0, stream>>>(bw, Wc, n16);

  // 2) Wc += sparse residual (fold sparse stream into the one fp16 GEMM)
  scatter_nnz_kernel<<<(nnz + 255) / 256, 256, 0, stream>>>(rows, cols, vals, Wc, nnz);

  // 3) x16 = fp16(x)
  const int n8 = (NROWS * IN_F) / 8;
  convert_x_kernel<<<(n8 + 255) / 256, 256, 0, stream>>>(x, x16, n8);

  // 4) out = x16 @ Wc^T  (fp32 accumulate via v_wmma_f32_16x16x32_f16)
  dim3 grid(OUT_F / BN, NROWS / BM);
  gemm_wmma_kernel<<<grid, 256, 0, stream>>>(x16, Wc, out);
}
